// ODConv2d_58574763983345
// MI455X (gfx1250) — matrix operand order repaired and checked by reading, not compiled, not run on hardware
//
#include <hip/hip_runtime.h>

typedef __attribute__((ext_vector_type(16))) _Float16 v16h;
typedef __attribute__((ext_vector_type(8)))  _Float16 v8h;
typedef __attribute__((ext_vector_type(8)))  float    v8f;

#define NB 16
#define NC 128
#define NH 96
#define NW 96
#define NPIX (NH*NW)          // 9216
#define NK 1152               // 9 * C   (K order: s*128 + c)
#define KN 4
#define ATT 16
#define CPAD 136              // padded channel stride in xRawT (bank-conflict free)

// ---------------------------------------------------------------------------
// 1) Global average pool: pooled[b*128+c] = mean over 96x96
// ---------------------------------------------------------------------------
__global__ __launch_bounds__(256)
void pool_kernel(const float* __restrict__ x, float* __restrict__ pooled) {
    __shared__ float red[256];
    const int bc = blockIdx.x;                       // b*128 + c
    const float* p = x + (size_t)bc * NPIX;
    float s = 0.f;
    for (int i = threadIdx.x; i < NPIX; i += 256) s += p[i];
    red[threadIdx.x] = s;
    __syncthreads();
    for (int off = 128; off > 0; off >>= 1) {
        if (threadIdx.x < off) red[threadIdx.x] += red[threadIdx.x + off];
        __syncthreads();
    }
    if (threadIdx.x == 0) pooled[bc] = red[0] * (1.f / (float)NPIX);
}

// ---------------------------------------------------------------------------
// 2) Attention MLP: hid -> ch_att (sigmoid), sp_att (sigmoid), k_att (softmax)
//    coef[b][k][s] = k_att[b][k] * sp_att[b][s]
// ---------------------------------------------------------------------------
__global__ __launch_bounds__(256)
void attn_kernel(const float* __restrict__ pooled,
                 const float* __restrict__ fc_w,
                 const float* __restrict__ cab_w, const float* __restrict__ cab_b,
                 const float* __restrict__ sab_w, const float* __restrict__ sab_b,
                 const float* __restrict__ kab_w, const float* __restrict__ kab_b,
                 float* __restrict__ ch_att, float* __restrict__ coef) {
    __shared__ float sp[NB * NC];
    __shared__ float sh[NB * ATT];
    __shared__ float ssp[NB * 9];
    __shared__ float sk[NB * KN];
    const int t = threadIdx.x;

    for (int i = t; i < NB * NC; i += 256) sp[i] = pooled[i];
    __syncthreads();

    {   // hid = relu(pooled @ fc_w^T) : 16x16 outputs, one per thread
        const int b = t >> 4, a = t & 15;
        float acc = 0.f;
        for (int c = 0; c < NC; ++c) acc += sp[b * NC + c] * fc_w[a * NC + c];
        sh[t] = acc > 0.f ? acc : 0.f;
    }
    __syncthreads();

    // ch_att = sigmoid(hid @ cab_w^T + cab_b) : 2048 outputs
    for (int i = t; i < NB * NC; i += 256) {
        const int b = i >> 7, c = i & 127;
        float acc = cab_b[c];
        for (int a = 0; a < ATT; ++a) acc += sh[b * ATT + a] * cab_w[c * ATT + a];
        ch_att[i] = 1.f / (1.f + __expf(-acc));
    }
    // sp_att = sigmoid(hid @ sab_w^T + sab_b) : 144 outputs
    if (t < NB * 9) {
        const int b = t / 9, s = t - b * 9;
        float acc = sab_b[s];
        for (int a = 0; a < ATT; ++a) acc += sh[b * ATT + a] * sab_w[s * ATT + a];
        ssp[t] = 1.f / (1.f + __expf(-acc));
    }
    // k raw logits : 64 outputs
    if (t < NB * KN) {
        const int b = t >> 2, k = t & 3;
        float acc = kab_b[k];
        for (int a = 0; a < ATT; ++a) acc += sh[b * ATT + a] * kab_w[k * ATT + a];
        sk[t] = acc;
    }
    __syncthreads();
    if (t < NB) {   // softmax over KN=4 per sample
        float m = sk[t * 4];
        for (int k = 1; k < 4; ++k) m = fmaxf(m, sk[t * 4 + k]);
        float e[4], sum = 0.f;
        for (int k = 0; k < 4; ++k) { e[k] = __expf(sk[t * 4 + k] - m); sum += e[k]; }
        const float inv = 1.f / sum;
        for (int k = 0; k < 4; ++k) sk[t * 4 + k] = e[k] * inv;
    }
    __syncthreads();
    for (int i = t; i < NB * KN * 9; i += 256) {
        const int b = i / 36, r = i - b * 36;
        const int k = r / 9, s = r - k * 9;
        coef[i] = sk[b * 4 + k] * ssp[b * 9 + s];
    }
}

// ---------------------------------------------------------------------------
// 3) Per-sample aggregated kernel, f16, K = s*128 + c (tap-major!):
//    aggW[(b*128+o)*1152 + s*128 + c] = sum_k coef[b][k][s] * weight[k][o][c][s]
//    weight index [k][o][c][ky][kx] = k*147456 + o*1152 + c*9 + s
// ---------------------------------------------------------------------------
__global__ __launch_bounds__(256)
void agg_kernel(const float* __restrict__ weight, const float* __restrict__ coef,
                _Float16* __restrict__ aggW) {
    __shared__ float sc[KN * 9];
    const int bo = blockIdx.x;                       // b*128 + o
    const int b = bo >> 7, o = bo & 127;
    if (threadIdx.x < KN * 9) sc[threadIdx.x] = coef[b * 36 + threadIdx.x];
    __syncthreads();
    const float* w0 = weight + o * NK;
    _Float16* dst = aggW + (size_t)bo * NK;
    for (int idx = threadIdx.x; idx < NK; idx += 256) {
        const int s = idx >> 7;                      // tap 0..8
        const int c = idx & 127;                     // channel
        float acc = 0.f;
#pragma unroll
        for (int k = 0; k < KN; ++k)
            acc += sc[k * 9 + s] * w0[k * (NC * NK) + c * 9 + s];
        dst[idx] = (_Float16)acc;
    }
}

// ---------------------------------------------------------------------------
// 4) Implicit-GEMM conv.  One block = one (sample, output row y).
//    M=128 (C_out), N=96 (x), K=1152 ordered (tap s, channel c).
//    Input staged ONCE per block: xRawT[3 rows][98 px][136 ch] f16 (80 KB),
//    ch_att fused, halo zeroed.  B fragments are direct 128-bit LDS loads.
//    8 waves = 4M x 2N; wave tile 32M x 48N -> 6 accumulators.
// ---------------------------------------------------------------------------
__global__ __launch_bounds__(256)
void conv_kernel(const float* __restrict__ x, const float* __restrict__ chAtt,
                 const _Float16* __restrict__ aggW, float* __restrict__ out) {
    __shared__ __align__(16) _Float16 xRawT[3 * 98 * CPAD];   // ~80 KB
    __shared__ __align__(16) _Float16 ldsA[128 * 32];         //   8 KB
    __shared__ float shCh[NC];

    const int b    = blockIdx.y;
    const int y0   = blockIdx.x;                    // output row 0..95
    const int t    = threadIdx.x;
    const int lane = t & 31;
    const int wave = t >> 5;
    const int wm   = wave >> 1;                     // 0..3 : M base = wm*32
    const int wn   = wave & 1;                      // 0..1 : N base = wn*48

    const float*    xb = x    + (size_t)b * NC * NPIX;
    const _Float16* aW = aggW + (size_t)b * NC * NK;

    if (t < NC) shCh[t] = chAtt[b * NC + t];
    __syncthreads();

    // ---- stage raw input: 128 ch x 3 rows x 96 px, float4 loads ----
    for (int i4 = t; i4 < NC * 3 * 24; i4 += 256) { // 24 float4 per row
        const int c   = i4 / 72;
        const int r   = i4 - c * 72;
        const int row = r / 24;                     // 0..2
        const int x4  = (r - row * 24) * 4;
        const int iy  = y0 + row - 1;
        float4 v = make_float4(0.f, 0.f, 0.f, 0.f);
        if ((unsigned)iy < (unsigned)NH)
            v = *(const float4*)(xb + c * NPIX + iy * NW + x4);
        const float sc = shCh[c];
        const int base = (row * 98 + 1 + x4) * CPAD + c;
        xRawT[base + 0 * CPAD] = (_Float16)(v.x * sc);
        xRawT[base + 1 * CPAD] = (_Float16)(v.y * sc);
        xRawT[base + 2 * CPAD] = (_Float16)(v.z * sc);
        xRawT[base + 3 * CPAD] = (_Float16)(v.w * sc);
    }
    // zero halo columns x=-1 and x=96
    for (int i = t; i < 3 * 2 * NC; i += 256) {
        const int c    = i & 127;
        const int rr   = i >> 7;                    // 0..5
        const int row  = rr >> 1;
        const int side = rr & 1;
        xRawT[(row * 98 + side * 97) * CPAD + c] = (_Float16)0.f;
    }

    v8f acc[2][3] = {};

    const int la  = lane & 15;
    const int kbA = (lane < 16) ? 0 : 8;    // A: K = [kbA,kbA+8) U [kbA+16,kbA+24)
    const int kbB = (lane < 16) ? 0 : 8;    // B: K = [kbB,kbB+8) U [kbB+16,kbB+24)

    // K loop: tap-major.  Each 32-step = one tap (dy,dx), 32 channels.
    for (int s = 0; s < 9; ++s) {
        const int dyrow    = s / 3;                          // 0..2 (scalar)
        const int dx       = s - dyrow * 3 - 1;              // -1..1
        const int rowBaseB = (dyrow * 98 + 1 + dx) * CPAD;   // + x*CPAD + c
        for (int cs = 0; cs < NC; cs += 32) {
            const int kk = s * NC + cs;
            // issue A-tile global loads before the barrier
            const int m  = t >> 1;
            const int ch = (t & 1) * 16;
            const v8h* srcA = (const v8h*)(aW + m * NK + kk + ch);
            const v8h a0 = srcA[0];
            const v8h a1 = srcA[1];
            if (kk + 32 < NK)
                __builtin_prefetch(aW + m * NK + kk + 32 + ch, 0, 0);
            __syncthreads();                 // prev iter done reading ldsA
            *(v8h*)&ldsA[m * 32 + ch]     = a0;
            *(v8h*)&ldsA[m * 32 + ch + 8] = a1;
            __syncthreads();

            // ---- fragments per ISA 16-bit layouts (§7.12.2) ----
            v16h aF[2], bF[3];
#pragma unroll
            for (int mi = 0; mi < 2; ++mi) {
                const int row = wm * 32 + mi * 16 + la;
                v8h f0 = *(const v8h*)&ldsA[row * 32 + kbA];
                v8h f1 = *(const v8h*)&ldsA[row * 32 + kbA + 16];
#pragma unroll
                for (int i = 0; i < 8; ++i) { aF[mi][i] = f0[i]; aF[mi][i + 8] = f1[i]; }
            }
#pragma unroll
            for (int ni = 0; ni < 3; ++ni) {
                const int xc = wn * 48 + ni * 16 + la;       // output x
                const _Float16* srcB = &xRawT[rowBaseB + xc * CPAD + cs + kbB];
                v8h f0 = *(const v8h*)(srcB);
                v8h f1 = *(const v8h*)(srcB + 16);
#pragma unroll
                for (int i = 0; i < 8; ++i) { bF[ni][i] = f0[i]; bF[ni][i + 8] = f1[i]; }
            }
#pragma unroll
            for (int mi = 0; mi < 2; ++mi)
#pragma unroll
                for (int ni = 0; ni < 3; ++ni)
                    acc[mi][ni] = __builtin_amdgcn_wmma_f32_16x16x32_f16(
                        false, aF[mi], false, bF[ni], (short)0, acc[mi][ni],
                        false, false);
        }
    }

    // ---- store D per C/D layout: M = j + 8*(lane>=16), N = lane&15 ----
    float* ob = out + (size_t)b * NC * NPIX + y0 * NW;
    const int mhi = (lane >> 4) * 8;
#pragma unroll
    for (int mi = 0; mi < 2; ++mi) {
#pragma unroll
        for (int ni = 0; ni < 3; ++ni) {
            const int col = wn * 48 + ni * 16 + (lane & 15);
#pragma unroll
            for (int j = 0; j < 8; ++j) {
                const int mrow = wm * 32 + mi * 16 + mhi + j;
                ob[(size_t)mrow * NPIX + col] = acc[mi][ni][j];
            }
        }
    }
}

// ---------------------------------------------------------------------------
extern "C" void kernel_launch(void* const* d_in, const int* in_sizes, int n_in,
                              void* d_out, int out_size, void* d_ws, size_t ws_size,
                              hipStream_t stream) {
    const float* x      = (const float*)d_in[0];
    const float* weight = (const float*)d_in[1];
    const float* fc_w   = (const float*)d_in[2];
    const float* cab_w  = (const float*)d_in[3];
    const float* cab_b  = (const float*)d_in[4];
    const float* sab_w  = (const float*)d_in[5];
    const float* sab_b  = (const float*)d_in[6];
    const float* kab_w  = (const float*)d_in[7];
    const float* kab_b  = (const float*)d_in[8];
    float* out = (float*)d_out;

    char* ws = (char*)d_ws;
    float*    pooled = (float*)(ws);             //  2048 f32 ->  8192 B
    float*    ch_att = (float*)(ws + 8192);      //  2048 f32 ->  8192 B
    float*    coef   = (float*)(ws + 16384);     //   576 f32 (pad to 4 KB)
    _Float16* aggW   = (_Float16*)(ws + 20480);  //  2359296 f16 -> 4.72 MB

    pool_kernel<<<NB * NC, 256, 0, stream>>>(x, pooled);
    attn_kernel<<<1, 256, 0, stream>>>(pooled, fc_w, cab_w, cab_b,
                                       sab_w, sab_b, kab_w, kab_b,
                                       ch_att, coef);
    agg_kernel<<<NB * NC, 256, 0, stream>>>(weight, coef, aggW);
    conv_kernel<<<dim3(NH, NB), 256, 0, stream>>>(x, ch_att, aggW, out);
}
